// CombLinearTCQ_6030134083824
// MI455X (gfx1250) — matrix operand order repaired and checked by reading, not compiled, not run on hardware
//
#include <hip/hip_runtime.h>

// ---------------------------------------------------------------------------
// CombLinearTCQ for MI455X (gfx1250, wave32):
//   phase 1: trellis decode -> bf16 weight matrix Wq[4096][4096] in d_ws
//   phase 2: bf16 WMMA GEMM  out[8192][4096] = x @ Wq^T  (fp32 accumulate)
// GEMM is compute-bound (275 GFLOP vs ~300MB traffic) -> v_wmma_f32_16x16x32_bf16.
// Wq (32MB) fits in the 192MB L2, so re-reads across the M dimension are cheap.
// ---------------------------------------------------------------------------

typedef __attribute__((ext_vector_type(8)))  float  v8f;
typedef __attribute__((ext_vector_type(8)))  __bf16 v8bf;
typedef __attribute__((ext_vector_type(16))) __bf16 v16bf;

#define GM 8192
#define GN 4096
#define GK 4096
#define BM 128
#define BN 128
#define BK 32
#define LDSW 40   // padded LDS row stride in halves (80B -> bank spread)

#ifndef __has_builtin
#define __has_builtin(x) 0
#endif

#if defined(__gfx1250__) && __has_builtin(__builtin_amdgcn_global_load_async_to_lds_b128)
#define USE_ASYNC_LDS 1
#else
#define USE_ASYNC_LDS 0
#endif

// Builtin prototype (from hipcc diagnostic): pointee is int __vector(4),
// AS1 (global) source, AS3 (LDS) destination.
typedef int v4i_t __attribute__((__vector_size__(16)));
typedef __attribute__((address_space(1))) v4i_t gbl_v4i;
typedef __attribute__((address_space(3))) v4i_t lds_v4i;

__device__ __forceinline__ unsigned short f2bf(float f) {
  unsigned int u = __float_as_uint(f);
  u += 0x7FFFu + ((u >> 16) & 1u);        // round-to-nearest-even
  return (unsigned short)(u >> 16);
}

// ---------------------------------------------------------------------------
// Trellis decode: one wave per 16x16 tile; lane handles steps t = lane + 32*s.
// Bitstream is MSB-first 16-bit words; window(t) = 16 bits starting at t*kv
// (circular). code = low 9 bits -> tlut[code] gives 2 floats at flat positions
// (2t, 2t+1) of the row-major 16x16 tile.
// ---------------------------------------------------------------------------
__global__ __launch_bounds__(256) void decode_kernel(
    const int* __restrict__ trellis, const float* __restrict__ tlut,
    unsigned short* __restrict__ Wq, int n_words, int kv, int m_off)
{
  const int wave = threadIdx.x >> 5;
  const int lane = threadIdx.x & 31;
  const int tile = blockIdx.x * 8 + wave;

  const int* words = trellis + (size_t)tile * n_words;
  const int mtile = tile >> 8;      // kt = 4096/16 = 256
  const int ktile = tile & 255;
  const int wmask = n_words - 1;    // n_words is 32 or 16 (pow2)

  const size_t out_base =
      (size_t)(m_off + mtile * 16) * GK + (size_t)ktile * 16;

#pragma unroll
  for (int s = 0; s < 4; ++s) {
    int t  = lane + s * 32;         // step 0..127
    int o  = t * kv;                // bit offset (< n_bits)
    int w  = o >> 4;
    int sh = o & 15;
    unsigned int c = (((unsigned)words[w] & 0xFFFFu) << 16) |
                     ((unsigned)words[(w + 1) & wmask] & 0xFFFFu);
    unsigned int window = (c >> (16 - sh)) & 0xFFFFu;   // sh==0 -> >>16 ok
    unsigned int code = window & 511u;                   // TLUT_BITS = 9
    float v0 = tlut[code * 2 + 0];
    float v1 = tlut[code * 2 + 1];
    int p   = t * 2;                 // even -> (v0,v1) share a row, col even
    int row = p >> 4;
    int col = p & 15;
    unsigned int packed = (unsigned)f2bf(v0) | ((unsigned)f2bf(v1) << 16);
    *(unsigned int*)(Wq + out_base + (size_t)row * GK + col) = packed;
  }
}

// ---------------------------------------------------------------------------
// GEMM: block tile 128x128, BK=32, 256 threads = 8 wave32s in a 2(M) x 4(N)
// grid; each wave owns a 64x32 output patch = 4x2 wmma 16x16 tiles.
// ---------------------------------------------------------------------------
__global__ __launch_bounds__(256, 2) void trellis_gemm(
    const float* __restrict__ X, const unsigned short* __restrict__ Wq,
    float* __restrict__ Out)
{
  __shared__ alignas(16) unsigned short As[BM * LDSW];
  __shared__ alignas(16) unsigned short Bs[BN * LDSW];

  const int tid  = threadIdx.x;
  const int wave = tid >> 5;
  const int lane = tid & 31;
  const int lh   = lane >> 4;    // lane half (0/1)
  const int l15  = lane & 15;

  const int mBase = blockIdx.y * BM;
  const int nBase = blockIdx.x * BN;
  const int wm = (wave & 1) * 64;   // wave M offset in block tile
  const int wn = (wave >> 1) * 32;  // wave N offset in block tile

  v8f acc[4][2] = {};

  for (int k0 = 0; k0 < GK; k0 += BK) {
    __syncthreads();   // previous iteration's fragment reads are done

    // ---- stage activations fp32 -> bf16 into LDS (128x32, 8 halves/chunk) --
#pragma unroll
    for (int it = 0; it < 2; ++it) {
      int c   = tid + it * 256;        // 0..511 chunks
      int row = c >> 2;
      int ck  = (c & 3) * 8;
      const float4* src =
          (const float4*)(X + (size_t)(mBase + row) * GK + k0 + ck);
      float4 f0 = src[0];
      float4 f1 = src[1];
      uint4 pk;
      pk.x = (unsigned)f2bf(f0.x) | ((unsigned)f2bf(f0.y) << 16);
      pk.y = (unsigned)f2bf(f0.z) | ((unsigned)f2bf(f0.w) << 16);
      pk.z = (unsigned)f2bf(f1.x) | ((unsigned)f2bf(f1.y) << 16);
      pk.w = (unsigned)f2bf(f1.z) | ((unsigned)f2bf(f1.w) << 16);
      *(uint4*)(As + row * LDSW + ck) = pk;
    }

    // ---- stage decoded bf16 weights into LDS (async-to-LDS when available) -
#pragma unroll
    for (int it = 0; it < 2; ++it) {
      int c   = tid + it * 256;
      int row = c >> 2;
      int ck  = (c & 3) * 8;
      size_t g = (size_t)(nBase + row) * GK + k0 + ck;
#if USE_ASYNC_LDS
      gbl_v4i* gp = (gbl_v4i*)(unsigned long long)(Wq + g);
      lds_v4i* lp =
          (lds_v4i*)(unsigned int)(unsigned long long)(Bs + row * LDSW + ck);
      __builtin_amdgcn_global_load_async_to_lds_b128(gp, lp, 0, 0);
#else
      uint4 d = *(const uint4*)(Wq + g);
      *(uint4*)(Bs + row * LDSW + ck) = d;
#endif
    }
#if USE_ASYNC_LDS
#if __has_builtin(__builtin_amdgcn_s_wait_asynccnt)
    __builtin_amdgcn_s_wait_asynccnt(0);
#else
    asm volatile("s_wait_asynccnt 0" ::: "memory");
#endif
#endif
    __syncthreads();

    // ---- build fragments per ISA 16-bit layouts -----------------------------
    // A (16x32, M on lanes 0..15): lane half lh reads K chunks
    //   {8*lh .. 8*lh+7} and {16+8*lh .. 16+8*lh+7}
    v16bf afrag[4];
#pragma unroll
    for (int i = 0; i < 4; ++i) {
      const unsigned short* base = As + (wm + i * 16 + l15) * LDSW + 8 * lh;
      v8bf lo = *(const v8bf*)(base);
      v8bf hi = *(const v8bf*)(base + 16);
      afrag[i] = __builtin_shufflevector(lo, hi, 0, 1, 2, 3, 4, 5, 6, 7, 8, 9,
                                         10, 11, 12, 13, 14, 15);
    }
    // B (32x16, N on lanes 0..15): lane half lh reads K chunks
    //   {8*lh .. 8*lh+7} and {16+8*lh .. 16+8*lh+7}
    v16bf bfrag[2];
#pragma unroll
    for (int j = 0; j < 2; ++j) {
      const unsigned short* base = Bs + (wn + j * 16 + l15) * LDSW + 8 * lh;
      v8bf lo = *(const v8bf*)(base);
      v8bf hi = *(const v8bf*)(base + 16);
      bfrag[j] = __builtin_shufflevector(lo, hi, 0, 1, 2, 3, 4, 5, 6, 7, 8, 9,
                                         10, 11, 12, 13, 14, 15);
    }

#pragma unroll
    for (int i = 0; i < 4; ++i)
#pragma unroll
      for (int j = 0; j < 2; ++j)
        acc[i][j] = __builtin_amdgcn_wmma_f32_16x16x32_bf16(
            false, afrag[i], false, bfrag[j], (short)0, acc[i][j], false,
            false);
  }

  // ---- epilogue: D layout — VGPR v holds M = v + 8*lh, N = lane&15 ---------
#pragma unroll
  for (int i = 0; i < 4; ++i) {
#pragma unroll
    for (int j = 0; j < 2; ++j) {
#pragma unroll
      for (int v = 0; v < 8; ++v) {
        int m = mBase + wm + i * 16 + v + 8 * lh;
        int n = nBase + wn + j * 16 + l15;
        Out[(size_t)m * GN + n] = acc[i][j][v];
      }
    }
  }
}

extern "C" void kernel_launch(void* const* d_in, const int* in_sizes, int n_in,
                              void* d_out, int out_size, void* d_ws,
                              size_t ws_size, hipStream_t stream) {
  const float* x    = (const float*)d_in[0];   // [8192, 4096] fp32
  const int*   t1   = (const int*)d_in[1];     // [32768, 32]
  const int*   t2   = (const int*)d_in[2];     // [32768, 16]
  const float* tlut = (const float*)d_in[3];   // [512, 2]
  float* out = (float*)d_out;                  // [8192, 4096] fp32
  unsigned short* Wq = (unsigned short*)d_ws;  // [4096, 4096] bf16 (32 MB)

  // decode: 32768 tiles each, 8 tiles (waves) per 256-thread block
  decode_kernel<<<4096, 256, 0, stream>>>(t1, tlut, Wq, 32, 4, 0);
  decode_kernel<<<4096, 256, 0, stream>>>(t2, tlut, Wq, 16, 2, 2048);

  dim3 grid(GN / BN, GM / BM);  // 32 x 64
  trellis_gemm<<<grid, 256, 0, stream>>>(x, Wq, out);
}
